// REGCN_28630251995290
// MI455X (gfx1250) — matrix operand order repaired and checked by reading, not compiled, not run on hardware
//
#include <hip/hip_runtime.h>

typedef __bf16 bf16;
typedef __attribute__((ext_vector_type(16))) __bf16 v16bf;
typedef __attribute__((ext_vector_type(8)))  float  v8f;

// ---- Problem constants (match reference setup_inputs()) ----
constexpr int N_NODES = 10000;
constexpr int N_EDGES = 320000;
constexpr int N_BASES = 30;
constexpr int D       = 200;

// ---- Padded GEMM geometry ----
constexpr int KP = 224;               // K (=D) padded to multiple of 32
constexpr int NY = N_BASES * D;       // 6000 message columns
constexpr int NLIVE = NY + D;         // 6200 live columns (bases + self)
constexpr int NP = 6208;              // N padded to multiple of 64 (97*64)
constexpr int MP = 10048;             // M padded to multiple of 64 (157*64)
constexpr int MBLK = MP / 64;         // 157
constexpr int NBLK = NP / 64;         // 97

// LDS tile row stride (bf16 elements): 32 data + 8 pad = 80 bytes (16B aligned)
constexpr int LDS_STRIDE = 40;

// ---- Workspace layout (bytes) ----
constexpr size_t OFF_A   = 0;                                        // bf16 [MP][KP]
constexpr size_t OFF_B   = OFF_A + (size_t)MP * KP * 2;              // bf16 [NP][KP]  (B^T!)
constexpr size_t OFF_Y   = OFF_B + (size_t)KP * NP * 2;              // bf16 [N_NODES][NY]
constexpr size_t OFF_AGG = OFF_Y + (size_t)N_NODES * NY * 2;         // f32  [N_NODES][D]
constexpr size_t OFF_DEG = OFF_AGG + (size_t)N_NODES * D * 4;        // f32  [N_NODES]

// Zero agg + deg (contiguous in workspace).
__global__ void rgcn_zero(float* __restrict__ p, int n) {
    int i = blockIdx.x * blockDim.x + threadIdx.x;
    if (i < n) p[i] = 0.0f;
}

// x (f32 [N_NODES][D]) -> padded bf16 A [MP][KP]
__global__ void rgcn_cvt_a(const float* __restrict__ x, bf16* __restrict__ A) {
    int i = blockIdx.x * blockDim.x + threadIdx.x;
    if (i >= MP * KP) return;
    int row = i / KP, k = i % KP;
    float v = (row < N_NODES && k < D) ? x[row * D + k] : 0.0f;
    A[i] = (bf16)v;
}

// weight (f32 [B][D][D]) + self_weight (f32 [D][D]) -> padded bf16 B^T [NP][KP]
// BT[n][k] = B[k][n]; col n < NY is basis b=n/D, out-dim d=n%D; cols [NY,NLIVE) = self.
__global__ void rgcn_cvt_b(const float* __restrict__ w, const float* __restrict__ sw,
                           bf16* __restrict__ BT) {
    int i = blockIdx.x * blockDim.x + threadIdx.x;
    if (i >= NP * KP) return;
    int n = i / KP, k = i % KP;
    float v = 0.0f;
    if (k < D) {
        if (n < NY) {
            int b = n / D, d = n % D;
            v = w[(b * D + k) * D + d];
        } else if (n < NLIVE) {
            v = sw[k * D + (n - NY)];
        }
    }
    BT[i] = (bf16)v;
}

// C(MP x NP) = A(MP x KP) * B(KP x NP) with B supplied transposed (BT = [NP][KP]).
// cols [0,NY) -> bf16 Y, cols [NY,NLIVE) -> out + bias.
// 256 threads = 8 wave32s arranged 4(M) x 2(N); each wave owns a 16x32 output slab
// of the 64x64 block tile; 2 WMMA per K-step, fragments built from b128 LDS loads.
__launch_bounds__(256, 2)
__global__ void rgcn_gemm(const bf16* __restrict__ A, const bf16* __restrict__ BT,
                          bf16* __restrict__ Y, const float* __restrict__ bias,
                          float* __restrict__ out) {
    __shared__ __align__(16) bf16 As[64 * LDS_STRIDE];   // [m][k]
    __shared__ __align__(16) bf16 Bs[64 * LDS_STRIDE];   // [n][k]

    const int tid  = threadIdx.x;
    const int lane = tid & 31;
    const int wave = tid >> 5;
    const int half = lane >> 4;      // 0: lanes 0-15, 1: lanes 16-31
    const int lr   = lane & 15;
    const int wm   = wave & 3;       // wave M tile (0..3)
    const int wn   = wave >> 2;      // wave N slab (0..1)
    const int m0   = blockIdx.x * 64;
    const int n0   = blockIdx.y * 64;

    // Global->LDS staging: 64 rows x 32 K-cols per tile, 16B (8 bf16) per thread.
    const int srow = tid >> 2;             // 0..63
    const int scol = (tid & 3) * 8;        // 0,8,16,24

    v8f c0 = {}; v8f c1 = {};

#pragma unroll
    for (int k0 = 0; k0 < KP; k0 += 32) {
        const uint4 av = *reinterpret_cast<const uint4*>(
            A + (size_t)(m0 + srow) * KP + k0 + scol);
        const uint4 bv = *reinterpret_cast<const uint4*>(
            BT + (size_t)(n0 + srow) * KP + k0 + scol);
        *reinterpret_cast<uint4*>(&As[srow * LDS_STRIDE + scol]) = av;
        *reinterpret_cast<uint4*>(&Bs[srow * LDS_STRIDE + scol]) = bv;
        __syncthreads();

        // A fragment (M = wm*16 + lr): element i holds K = half*16 + i
        // -> two b128 LDS loads.
        v16bf a;
        {
            uint4* ap = reinterpret_cast<uint4*>(&a);
            const uint4* asr =
                reinterpret_cast<const uint4*>(&As[(wm * 16 + lr) * LDS_STRIDE]);
            ap[0] = asr[half * 2];
            ap[1] = asr[half * 2 + 1];
        }
        // B fragments (N = wn*32 + lr, +16): element i holds K = half*16 + i
        // -> 32 contiguous bytes of the BT row -> two b128 LDS loads each.
        v16bf b0, b1;
        {
            uint4* bp0 = reinterpret_cast<uint4*>(&b0);
            uint4* bp1 = reinterpret_cast<uint4*>(&b1);
            const uint4* bsr0 =
                reinterpret_cast<const uint4*>(&Bs[(wn * 32 + lr) * LDS_STRIDE]);
            const uint4* bsr1 =
                reinterpret_cast<const uint4*>(&Bs[(wn * 32 + lr + 16) * LDS_STRIDE]);
            bp0[0] = bsr0[half * 2];
            bp0[1] = bsr0[half * 2 + 1];
            bp1[0] = bsr1[half * 2];
            bp1[1] = bsr1[half * 2 + 1];
        }
        c0 = __builtin_amdgcn_wmma_f32_16x16x32_bf16(false, a, false, b0,
                                                     (short)0, c0, false, false);
        c1 = __builtin_amdgcn_wmma_f32_16x16x32_bf16(false, a, false, b1,
                                                     (short)0, c1, false, false);
        __syncthreads();
    }

    // D layout: VGPR r -> M = r + 8*half, N = lr.
#pragma unroll
    for (int r = 0; r < 8; ++r) {
        const int m = m0 + wm * 16 + half * 8 + r;
        if (m >= N_NODES) continue;
        const int nA = n0 + wn * 32 + lr;
        const int nB = nA + 16;
        const float vA = c0[r];
        const float vB = c1[r];
        if (nA < NY)          Y[(size_t)m * NY + nA] = (bf16)vA;
        else if (nA < NLIVE)  out[m * D + (nA - NY)] = vA + bias[nA - NY];
        if (nB < NY)          Y[(size_t)m * NY + nB] = (bf16)vB;
        else if (nB < NLIVE)  out[m * D + (nB - NY)] = vB + bias[nB - NY];
    }
}

// One block per edge: msg[d] = sum_b coeff[type,b] * Y[src, b, d]; atomic add into agg[tgt].
__launch_bounds__(256)
__global__ void rgcn_scatter(const bf16* __restrict__ Y, const float* __restrict__ coeff,
                             const int* __restrict__ ei, const int* __restrict__ et,
                             float* __restrict__ agg, float* __restrict__ deg) {
    __shared__ float lc[N_BASES];
    const int e   = blockIdx.x;
    const int src = ei[e];
    const int tgt = ei[N_EDGES + e];
    const int rel = et[e];
    const bf16* yrow = Y + (size_t)src * NY;
    if (threadIdx.x == 0) __builtin_prefetch(yrow, 0, 0);   // global_prefetch_b8
    if (threadIdx.x < N_BASES) lc[threadIdx.x] = coeff[rel * N_BASES + threadIdx.x];
    __syncthreads();
    const int d = threadIdx.x;
    if (d == 0) atomicAdd(&deg[tgt], 1.0f);
    if (d < D) {
        float acc = 0.0f;
#pragma unroll
        for (int b = 0; b < N_BASES; ++b)
            acc += lc[b] * (float)yrow[b * D + d];
        atomicAdd(&agg[tgt * D + d], acc);
    }
}

// out += agg / max(deg, 1)   (out already holds x@W_self + bias from the GEMM epilogue)
__global__ void rgcn_finalize(const float* __restrict__ agg, const float* __restrict__ deg,
                              float* __restrict__ out) {
    int i = blockIdx.x * blockDim.x + threadIdx.x;
    if (i < N_NODES * D) {
        float dg = deg[i / D];
        out[i] += agg[i] / (dg > 1.0f ? dg : 1.0f);
    }
}

extern "C" void kernel_launch(void* const* d_in, const int* in_sizes, int n_in,
                              void* d_out, int out_size, void* d_ws, size_t ws_size,
                              hipStream_t stream) {
    const float* x     = (const float*)d_in[0];
    const float* w     = (const float*)d_in[1];
    const float* coeff = (const float*)d_in[2];
    const float* sw    = (const float*)d_in[3];
    const float* bias  = (const float*)d_in[4];
    const int*   ei    = (const int*)d_in[5];
    const int*   et    = (const int*)d_in[6];
    float*       out   = (float*)d_out;

    char* ws  = (char*)d_ws;
    bf16*  A   = (bf16*)(ws + OFF_A);
    bf16*  BT  = (bf16*)(ws + OFF_B);
    bf16*  Y   = (bf16*)(ws + OFF_Y);
    float* agg = (float*)(ws + OFF_AGG);
    float* deg = (float*)(ws + OFF_DEG);

    {   // zero agg + deg (contiguous)
        int n = N_NODES * D + N_NODES;
        rgcn_zero<<<(n + 255) / 256, 256, 0, stream>>>(agg, n);
    }
    rgcn_cvt_a<<<(MP * KP + 255) / 256, 256, 0, stream>>>(x, A);
    rgcn_cvt_b<<<(NP * KP + 255) / 256, 256, 0, stream>>>(w, sw, BT);
    rgcn_gemm<<<dim3(MBLK, NBLK), 256, 0, stream>>>(A, BT, Y, bias, out);
    rgcn_scatter<<<N_EDGES, 256, 0, stream>>>(Y, coeff, ei, et, agg, deg);
    rgcn_finalize<<<(N_NODES * D + 255) / 256, 256, 0, stream>>>(agg, deg, out);
}
